// MultiHeadedAttention_40690520162562
// MI455X (gfx1250) — matrix operand order repaired and checked by reading, not compiled, not run on hardware
//
#include <hip/hip_runtime.h>

// ---------------- types ----------------
typedef __attribute__((ext_vector_type(16))) _Float16 v16h;
typedef __attribute__((ext_vector_type(8)))  _Float16 v8h;
typedef __attribute__((ext_vector_type(8)))  float    v8f;

#define WMMA_F16(a, b, c) \
    __builtin_amdgcn_wmma_f32_16x16x32_f16(false, (a), false, (b), (short)0, (c), false, false)

// Problem constants
#define BATCH 2
#define SEQ   2048
#define HEADS 8
#define DK    64
#define DM    512
#define MROWS (BATCH * SEQ)   // 4096

// ---------------- workspace layout (bytes) ----------------
#define MB (1u << 20)
#define OFF_XQ   (0u)                 // 4 MiB  f16 query
#define OFF_XK   (4u * MB)            // 4 MiB  f16 key
#define OFF_XV   (8u * MB)            // 4 MiB  f16 value
#define OFF_WQ   (12u * MB)           // 512 KiB
#define OFF_WK   (12u * MB + 512u * 1024u)
#define OFF_WV   (13u * MB)
#define OFF_WO   (13u * MB + 512u * 1024u)
#define OFF_Q    (14u * MB)           // 4 MiB  [B,H,S,64] f16
#define OFF_K    (18u * MB)           // 4 MiB  [B,H,S,64] f16
#define OFF_VT   (22u * MB)           // 4 MiB  [B,H,64,S] f16
#define OFF_CTX  (26u * MB)           // 4 MiB  [B,S,512]  f16

// ---------------- fragment loaders (ISA 7.12.2 wave32 layouts) ----------------
// A 16x32 f16: lane(0-15) row=lane, elems 0..7 -> K=k0+hi*8..+7, elems 8..15 -> K=k0+16+hi*8..+7
static __device__ inline v16h load_a_frag(const _Float16* __restrict__ rowptr, int k0, int hi) {
    v8h a0 = *(const v8h*)(rowptr + k0 + hi * 8);
    v8h a1 = *(const v8h*)(rowptr + k0 + 16 + hi * 8);
    v16h a;
#pragma unroll
    for (int i = 0; i < 8; ++i) { a[i] = a0[i]; a[i + 8] = a1[i]; }
    return a;
}

// B 32x16 f16: lane holds column (lane&15); elems 0..7 -> K=k0+hi*8..+7, elems 8..15 -> K=k0+16+hi*8..+7
// `colptr` points at the source row that represents this lane's B column.
static __device__ inline v16h load_b_frag(const _Float16* __restrict__ colptr, int k0, int hi) {
    v8h b0 = *(const v8h*)(colptr + k0 + hi * 8);
    v8h b1 = *(const v8h*)(colptr + k0 + 16 + hi * 8);
    v16h b;
#pragma unroll
    for (int i = 0; i < 8; ++i) { b[i] = b0[i]; b[i + 8] = b1[i]; }
    return b;
}

// 16-lane-group reductions (xor masks 1,2,4,8 stay within lanes 0-15 / 16-31)
static __device__ inline float rowmax16(float v) {
    v = fmaxf(v, __shfl_xor(v, 1, 32));
    v = fmaxf(v, __shfl_xor(v, 2, 32));
    v = fmaxf(v, __shfl_xor(v, 4, 32));
    v = fmaxf(v, __shfl_xor(v, 8, 32));
    return v;
}
static __device__ inline float rowsum16(float v) {
    v += __shfl_xor(v, 1, 32);
    v += __shfl_xor(v, 2, 32);
    v += __shfl_xor(v, 4, 32);
    v += __shfl_xor(v, 8, 32);
    return v;
}

// ---------------- kernels ----------------
__global__ void f32_to_f16_kernel(const float* __restrict__ in, _Float16* __restrict__ out, int n) {
    int i = blockIdx.x * blockDim.x + threadIdx.x;
    if (i < n) out[i] = (_Float16)in[i];
}

// C[m,n] = sum_k X[m,k] * W[n,k] + bias[n], 32x32 output tile per wave (2x2 WMMA blocking).
// mode 0: store f16 to [B,H,S,64] (Q)     mode 1: same (K)
// mode 2: store f16 to [B,H,64,S] (V^T)   mode 3: store f32 row-major [M,512] (output proj)
__global__ __launch_bounds__(128) void gemm_xwt_wmma(const _Float16* __restrict__ X,
                                                     const _Float16* __restrict__ W,
                                                     const float* __restrict__ bias,
                                                     void* __restrict__ outp, int mode) {
    const int wave = threadIdx.x >> 5;
    const int lane = threadIdx.x & 31;
    const int t    = blockIdx.x * 4 + wave;     // 2048 tiles of 32x32
    const int tn   = t & 15;                    // 512/32 = 16 col supertiles
    const int tm   = t >> 4;                    // 4096/32 = 128 row supertiles
    const int m0 = tm * 32, n0 = tn * 32;
    const int rc = lane & 15, hi = lane >> 4;   // rc = A-row / B-col for this lane

    const _Float16* xrow0 = X + (size_t)(m0 + rc) * DM;
    const _Float16* xrow1 = X + (size_t)(m0 + 16 + rc) * DM;
    const _Float16* wrow0 = W + (size_t)(n0 + rc) * DM;
    const _Float16* wrow1 = W + (size_t)(n0 + 16 + rc) * DM;

    v8f c00 = {}, c01 = {}, c10 = {}, c11 = {};
#pragma unroll 2
    for (int k0 = 0; k0 < DM; k0 += 32) {
        v16h a0 = load_a_frag(xrow0, k0, hi);
        v16h a1 = load_a_frag(xrow1, k0, hi);
        v16h b0 = load_b_frag(wrow0, k0, hi);
        v16h b1 = load_b_frag(wrow1, k0, hi);
        c00 = WMMA_F16(a0, b0, c00);
        c01 = WMMA_F16(a0, b1, c01);
        c10 = WMMA_F16(a1, b0, c10);
        c11 = WMMA_F16(a1, b1, c11);
    }

    v8f ctile[4] = {c00, c01, c10, c11};
#pragma unroll
    for (int ti = 0; ti < 2; ++ti) {
#pragma unroll
        for (int tj = 0; tj < 2; ++tj) {
            const v8f c  = ctile[ti * 2 + tj];
            const int nc = n0 + tj * 16 + rc;
            const float bn = bias[nc];
#pragma unroll
            for (int r = 0; r < 8; ++r) {
                const int m   = m0 + ti * 16 + r + hi * 8;
                const float v = c[r] + bn;
                if (mode == 3) {
                    ((float*)outp)[(size_t)m * DM + nc] = v;
                } else {
                    const int bb = m >> 11, s = m & (SEQ - 1);
                    const int h = nc >> 6, d = nc & 63;
                    size_t idx = (mode == 2)
                        ? ((size_t)((bb * HEADS + h) * DK + d) * SEQ + s)      // V^T
                        : ((size_t)((bb * HEADS + h) * SEQ + s) * DK + d);     // Q / K
                    ((_Float16*)outp)[idx] = (_Float16)v;
                }
            }
        }
    }
}

// Flash attention: one wave per 16-query tile; 32-key blocks; diag mask; online softmax.
__global__ __launch_bounds__(128) void attn_wmma(const _Float16* __restrict__ Q,
                                                 const _Float16* __restrict__ K,
                                                 const _Float16* __restrict__ Vt,
                                                 _Float16* __restrict__ ctx) {
    __shared__ _Float16 plds[4][16 * 32];       // per-wave P staging (C-layout -> A-layout)
    const int wave = threadIdx.x >> 5;
    const int lane = threadIdx.x & 31;
    const int g  = blockIdx.x * 4 + wave;       // 2048 waves total
    const int qt = g & 127, bh = g >> 7;
    const int b = bh >> 3, h = bh & 7;
    const int q0 = qt * 16;
    const int rc = lane & 15, hi = lane >> 4;

    const _Float16* Qb = Q  + (size_t)bh * SEQ * DK;
    const _Float16* Kb = K  + (size_t)bh * SEQ * DK;
    const _Float16* Vb = Vt + (size_t)bh * DK * SEQ;

    const _Float16* qrow = Qb + (size_t)(q0 + rc) * DK;
    v16h qa0 = load_a_frag(qrow, 0,  hi);
    v16h qa1 = load_a_frag(qrow, 32, hi);

    float mrow[8], lrow[8];
    v8f acc[4];
#pragma unroll
    for (int r = 0; r < 8; ++r) { mrow[r] = -3.0e38f; lrow[r] = 0.0f; }
#pragma unroll
    for (int t = 0; t < 4; ++t) acc[t] = (v8f){};

    _Float16* pl = &plds[wave][0];

    for (int j = 0; j < SEQ / 32; ++j) {
        const int kb = j * 32;
        const _Float16* k0r = Kb + (size_t)(kb + rc) * DK;
        const _Float16* k1r = Kb + (size_t)(kb + 16 + rc) * DK;

        v8f s0 = {}, s1 = {};
        s0 = WMMA_F16(qa0, load_b_frag(k0r, 0,  hi), s0);
        s0 = WMMA_F16(qa1, load_b_frag(k0r, 32, hi), s0);
        s1 = WMMA_F16(qa0, load_b_frag(k1r, 0,  hi), s1);
        s1 = WMMA_F16(qa1, load_b_frag(k1r, 32, hi), s1);

        // scale 1/sqrt(64) and diagonal mask; online softmax update
#pragma unroll
        for (int r = 0; r < 8; ++r) {
            const int qg = q0 + r + hi * 8;
            float v0 = s0[r] * 0.125f;
            float v1 = s1[r] * 0.125f;
            if (qg == kb + rc)      v0 = -1.0e9f;
            if (qg == kb + 16 + rc) v1 = -1.0e9f;

            const float rmax = rowmax16(fmaxf(v0, v1));
            const float mn   = fmaxf(mrow[r], rmax);
            const float sc   = __expf(mrow[r] - mn);
            mrow[r] = mn;
            const float p0 = __expf(v0 - mn);
            const float p1 = __expf(v1 - mn);
            lrow[r] = lrow[r] * sc + rowsum16(p0 + p1);
#pragma unroll
            for (int t = 0; t < 4; ++t) acc[t][r] *= sc;
            s0[r] = p0;
            s1[r] = p1;
        }

        // C-layout -> A-layout via per-wave LDS (DS ops are in-order within a wave)
        __builtin_amdgcn_wave_barrier();
#pragma unroll
        for (int r = 0; r < 8; ++r) {
            const int rowi = r + hi * 8;
            pl[rowi * 32 + rc]      = (_Float16)s0[r];
            pl[rowi * 32 + 16 + rc] = (_Float16)s1[r];
        }
        __builtin_amdgcn_wave_barrier();
        asm volatile("s_wait_dscnt 0" ::: "memory");
        v16h pa = load_a_frag(pl + rc * 32, 0, hi);
        __builtin_amdgcn_wave_barrier();

        // P (16x32) @ V (32x64): B cols are d = t*16 + rc, contiguous over keys in V^T
#pragma unroll
        for (int t = 0; t < 4; ++t) {
            const _Float16* vrow = Vb + (size_t)(t * 16 + rc) * SEQ;
            acc[t] = WMMA_F16(pa, load_b_frag(vrow, kb, hi), acc[t]);
        }
    }

    // normalize and write context [B,S,512] f16
    float inv[8];
#pragma unroll
    for (int r = 0; r < 8; ++r) inv[r] = 1.0f / lrow[r];
#pragma unroll
    for (int t = 0; t < 4; ++t) {
#pragma unroll
        for (int r = 0; r < 8; ++r) {
            const int s = q0 + r + hi * 8;
            const int d = h * DK + t * 16 + rc;
            ctx[((size_t)b * SEQ + s) * DM + d] = (_Float16)(acc[t][r] * inv[r]);
        }
    }
}

// ---------------- host ----------------
extern "C" void kernel_launch(void* const* d_in, const int* in_sizes, int n_in,
                              void* d_out, int out_size, void* d_ws, size_t ws_size,
                              hipStream_t stream) {
    (void)in_sizes; (void)n_in; (void)out_size; (void)ws_size;
    const float* q  = (const float*)d_in[0];
    const float* k  = (const float*)d_in[1];
    const float* v  = (const float*)d_in[2];
    const float* Wq = (const float*)d_in[3];
    const float* bq = (const float*)d_in[4];
    const float* Wk = (const float*)d_in[5];
    const float* bk = (const float*)d_in[6];
    const float* Wv = (const float*)d_in[7];
    const float* bv = (const float*)d_in[8];
    const float* Wo = (const float*)d_in[9];
    const float* bo = (const float*)d_in[10];

    char* ws = (char*)d_ws;
    _Float16* Xq  = (_Float16*)(ws + OFF_XQ);
    _Float16* Xk  = (_Float16*)(ws + OFF_XK);
    _Float16* Xv  = (_Float16*)(ws + OFF_XV);
    _Float16* Wqh = (_Float16*)(ws + OFF_WQ);
    _Float16* Wkh = (_Float16*)(ws + OFF_WK);
    _Float16* Wvh = (_Float16*)(ws + OFF_WV);
    _Float16* Woh = (_Float16*)(ws + OFF_WO);
    _Float16* Qh  = (_Float16*)(ws + OFF_Q);
    _Float16* Kh  = (_Float16*)(ws + OFF_K);
    _Float16* Vth = (_Float16*)(ws + OFF_VT);
    _Float16* Ctx = (_Float16*)(ws + OFF_CTX);

    const int nX = MROWS * DM;   // 2,097,152
    const int nW = DM * DM;      //   262,144
    f32_to_f16_kernel<<<(nX + 255) / 256, 256, 0, stream>>>(q,  Xq,  nX);
    f32_to_f16_kernel<<<(nX + 255) / 256, 256, 0, stream>>>(k,  Xk,  nX);
    f32_to_f16_kernel<<<(nX + 255) / 256, 256, 0, stream>>>(v,  Xv,  nX);
    f32_to_f16_kernel<<<(nW + 255) / 256, 256, 0, stream>>>(Wq, Wqh, nW);
    f32_to_f16_kernel<<<(nW + 255) / 256, 256, 0, stream>>>(Wk, Wkh, nW);
    f32_to_f16_kernel<<<(nW + 255) / 256, 256, 0, stream>>>(Wv, Wvh, nW);
    f32_to_f16_kernel<<<(nW + 255) / 256, 256, 0, stream>>>(Wo, Woh, nW);

    // 2048 32x32 supertiles, 4 waves/block
    gemm_xwt_wmma<<<512, 128, 0, stream>>>(Xq, Wqh, bq, Qh,  0);
    gemm_xwt_wmma<<<512, 128, 0, stream>>>(Xk, Wkh, bk, Kh,  1);
    gemm_xwt_wmma<<<512, 128, 0, stream>>>(Xv, Wvh, bv, Vth, 2);

    // 2048 query-tile waves, 4 waves/block
    attn_wmma<<<512, 128, 0, stream>>>(Qh, Kh, Vth, Ctx);

    // output projection -> f32
    gemm_xwt_wmma<<<512, 128, 0, stream>>>(Ctx, Woh, bo, d_out, 3);
}
